// PointNetSetAbstractionMsg_24120536335103
// MI455X (gfx1250) — matrix operand order repaired and checked by reading, not compiled, not run on hardware
//
#include <hip/hip_runtime.h>
#include <hip/hip_bf16.h>

// PointNet++ MSG set abstraction for gfx1250 (MI455X).
// fps -> ball query (3 radii) -> per branch:
//   gather+layer0 (scalar, Cin=6) -> stats -> WMMA layer1 -> stats ->
//   WMMA layer2 -> stats -> maxpool(norm+relu) -> d_out.
// Activations stored pre-BN as f16 (stage is HBM-bound: ~1.5 GB traffic at
// 23.3 TB/s dominates the ~53 GFLOP of matmul); normalize+ReLU fused into the
// consumer's LDS staging; matmuls use v_wmma_f32_16x16x32_f16 with A and B
// fragments both loaded as contiguous 16B LDS vectors.

typedef _Float16 h16;
typedef _Float16 v16h __attribute__((ext_vector_type(16)));
typedef _Float16 v8h  __attribute__((ext_vector_type(8)));
typedef float    v8f  __attribute__((ext_vector_type(8)));

#define BATCH 16
#define NPTS  4096
#define SOUT  512
#define EPSI  1e-5f

// ---------------------------------------------------------------- FPS ----
__global__ void fps_kernel(const float* __restrict__ xyz, int* __restrict__ cent,
                           float* __restrict__ newxyz, float* __restrict__ out_xyz)
{
    const int b   = blockIdx.x;
    const int tid = threadIdx.x;                 // 256 threads
    const float* xb = xyz + b * 3 * NPTS;
    const int PPT = NPTS / 256;                  // 16 points per thread
    float dist[PPT];
#pragma unroll
    for (int i = 0; i < PPT; ++i) dist[i] = 1e10f;

    __shared__ int   s_far;
    __shared__ float rv[256];
    __shared__ int   ri[256];
    if (tid == 0) s_far = 0;
    __syncthreads();

    for (int it = 0; it < SOUT; ++it) {
        int far = s_far;
        if (tid == 0) cent[b * SOUT + it] = far;
        float cx = xb[far], cy = xb[NPTS + far], cz = xb[2 * NPTS + far];
        float best = -1.0f; int bidx = 0;
#pragma unroll
        for (int i = 0; i < PPT; ++i) {
            int n = tid + i * 256;
            float dx = xb[n] - cx, dy = xb[NPTS + n] - cy, dz = xb[2 * NPTS + n] - cz;
            float d  = dx * dx + dy * dy + dz * dz;
            float dm = fminf(dist[i], d);
            dist[i] = dm;
            if (dm > best || (dm == best && n < bidx)) { best = dm; bidx = n; }
        }
        rv[tid] = best; ri[tid] = bidx;
        __syncthreads();
        for (int off = 128; off > 0; off >>= 1) {
            if (tid < off) {
                float ov = rv[tid + off]; int oi = ri[tid + off];
                if (ov > rv[tid] || (ov == rv[tid] && oi < ri[tid])) { rv[tid] = ov; ri[tid] = oi; }
            }
            __syncthreads();
        }
        if (tid == 0) s_far = ri[0];
        __syncthreads();
    }
    // gather centroids: new_xyz (ws, point-major) + d_out part 1 (B,3,S)
    for (int s = tid; s < SOUT; s += 256) {
        int gi = cent[b * SOUT + s];
#pragma unroll
        for (int c = 0; c < 3; ++c) {
            float v = xb[c * NPTS + gi];
            newxyz[(b * SOUT + s) * 3 + c] = v;
            out_xyz[b * 3 * SOUT + c * SOUT + s] = v;
        }
    }
}

// ---------------------------------------------------------- ball query ----
__global__ void ballquery_kernel(const float* __restrict__ xyz,
                                 const float* __restrict__ newxyz,
                                 int* __restrict__ g0, int* __restrict__ g1,
                                 int* __restrict__ g2)
{
    const int wave = blockIdx.x * (blockDim.x >> 5) + (threadIdx.x >> 5);
    const int lane = threadIdx.x & 31;
    if (wave >= BATCH * SOUT) return;
    const int b = wave / SOUT;
    const float* xb = xyz + b * 3 * NPTS;
    const float cx = newxyz[wave * 3 + 0];
    const float cy = newxyz[wave * 3 + 1];
    const float cz = newxyz[wave * 3 + 2];

    const float r2[3] = {0.01f, 0.04f, 0.16f};
    const int   Ks[3] = {16, 32, 128};
    int* gp[3];
    gp[0] = g0 + wave * 16;
    gp[1] = g1 + wave * 32;
    gp[2] = g2 + wave * 128;
    int cnt[3]   = {0, 0, 0};
    int first[3] = {-1, -1, -1};

    for (int n0 = 0; n0 < NPTS; n0 += 32) {
        int n = n0 + lane;
        float dx = xb[n] - cx, dy = xb[NPTS + n] - cy, dz = xb[2 * NPTS + n] - cz;
        float d = dx * dx + dy * dy + dz * dz;
#pragma unroll
        for (int r = 0; r < 3; ++r) {
            if (cnt[r] >= Ks[r]) continue;               // wave-uniform
            unsigned m = (unsigned)__ballot(d <= r2[r]);
            if (m) {
                if (first[r] < 0) first[r] = n0 + __ffs((int)m) - 1;
                int pos = cnt[r] + __popc(m & ((1u << lane) - 1u));
                if (((m >> lane) & 1u) && pos < Ks[r]) gp[r][pos] = n;
                cnt[r] += __popc(m);
            }
        }
        if (cnt[0] >= 16 && cnt[1] >= 32 && cnt[2] >= 128) break;
    }
#pragma unroll
    for (int r = 0; r < 3; ++r) {
        int f = first[r] < 0 ? 0 : first[r];
        int start = cnt[r] < Ks[r] ? cnt[r] : Ks[r];
        for (int pos = start + lane; pos < Ks[r]; pos += 32) gp[r][pos] = f;
    }
}

// --------------------------------------------- gather + layer0 (Cin=6) ----
__global__ void gather_l0_kernel(const float* __restrict__ xyz, const float* __restrict__ pts,
                                 const float* __restrict__ newxyz, const int* __restrict__ gidx,
                                 const float* __restrict__ W, const float* __restrict__ bias,
                                 h16* __restrict__ actOut, int K, int C0, int P)
{
    __shared__ float sW[64 * 6];
    __shared__ float sB[64];
    for (int i = threadIdx.x; i < C0 * 6; i += blockDim.x) sW[i] = W[i];
    for (int i = threadIdx.x; i < C0;     i += blockDim.x) sB[i] = bias[i];
    __syncthreads();

    int p = blockIdx.x * blockDim.x + threadIdx.x;
    if (p >= P) return;
    int g = p / K, k = p - g * K;
    int b = g / SOUT;
    int gi = gidx[g * K + k];
    const float* xb = xyz + b * 3 * NPTS;
    const float* pb = pts + b * 3 * NPTS;
    float f[6];
    f[0] = pb[gi]; f[1] = pb[NPTS + gi]; f[2] = pb[2 * NPTS + gi];
    f[3] = xb[gi]            - newxyz[g * 3 + 0];
    f[4] = xb[NPTS + gi]     - newxyz[g * 3 + 1];
    f[5] = xb[2 * NPTS + gi] - newxyz[g * 3 + 2];
    for (int o = 0; o < C0; ++o) {
        float acc = sB[o];
#pragma unroll
        for (int c = 0; c < 6; ++c) acc += sW[o * 6 + c] * f[c];
        actOut[o * P + p] = (h16)acc;                 // int indexing: SALU mul
    }
}

// ----------------------------------------------------- BN stat kernels ----
// stats layout per layer (stride 128): [0..C) sum, [128..) sumsq, [256..) mean, [384..) rstd
__global__ void zero_stats_kernel(float* __restrict__ stats, int nfloats)
{
    for (int i = blockIdx.x * blockDim.x + threadIdx.x; i < nfloats; i += blockDim.x * gridDim.x)
        stats[i] = 0.0f;
}

__global__ void stats_kernel(const h16* __restrict__ act, float* __restrict__ stats,
                             int P, int bpc)
{
    const int CHUNK = 32768;                      // elements; multiple of 8*256
    int ch  = blockIdx.x / bpc;
    int blk = blockIdx.x % bpc;
    int lo = blk * CHUNK;
    int hi = lo + CHUNK < P ? lo + CHUNK : P;     // P always multiple of 8
    const v8h* a8 = (const v8h*)(act + ch * P);
    float s = 0.0f, ss = 0.0f;
    for (int i = lo / 8 + threadIdx.x; i < hi / 8; i += blockDim.x) {
        v8h v = a8[i];
#pragma unroll
        for (int e = 0; e < 8; ++e) { float f = (float)v[e]; s += f; ss += f * f; }
    }
    __shared__ float rs[256], rss[256];
    rs[threadIdx.x] = s; rss[threadIdx.x] = ss;
    __syncthreads();
    for (int off = 128; off > 0; off >>= 1) {
        if (threadIdx.x < off) {
            rs[threadIdx.x]  += rs[threadIdx.x + off];
            rss[threadIdx.x] += rss[threadIdx.x + off];
        }
        __syncthreads();
    }
    if (threadIdx.x == 0) {
        atomicAdd(&stats[ch], rs[0]);
        atomicAdd(&stats[128 + ch], rss[0]);
    }
}

__global__ void finalize_kernel(float* __restrict__ stats, int C, float invP)
{
    int c = threadIdx.x;
    if (c < C) {
        float mean = stats[c] * invP;
        float var  = stats[128 + c] * invP - mean * mean;
        stats[256 + c] = mean;
        stats[384 + c] = rsqrtf(var + EPSI);
    }
}

// --------------------------------------------------- WMMA conv layer ------
// D[Cout,P] = W[Cout,Cin] x relu(norm(actIn))[Cin,P] + b ; pre-BN f16 out.
// 128-position tile per block; 4 waves; X staged transposed [pos][ch] so the
// B fragment is two 16B runs per lane (2x ds_load_b128), same as A.
#define NTILE 128
template <int CIN, int COUT>
__global__ void __launch_bounds__(128)
mlp_wmma_kernel(const h16* __restrict__ actIn, h16* __restrict__ actOut,
                const float* __restrict__ W, const float* __restrict__ bias,
                const float* __restrict__ inStats,   // producer layer mean/rstd
                const float* __restrict__ inGamma, const float* __restrict__ inBeta,
                int P)
{
    constexpr int LDK = CIN + 8;                  // stride in halves; 16B multiple
    __shared__ h16 sW[COUT * LDK];                // [Cout][Cin]
    __shared__ h16 sXT[NTILE * LDK];              // [pos][Cin], normalized+relu
    const int tid  = threadIdx.x;
    const int lane = tid & 31;
    const int wv   = tid >> 5;
    const int pBase = blockIdx.x * NTILE;         // P is a multiple of NTILE

    // stage W -> f16 LDS (paired: b64 global load, b32 LDS store)
    for (int i2 = tid; i2 < COUT * (CIN / 2); i2 += 128) {
        int o = i2 / (CIN / 2);
        int c = (i2 - o * (CIN / 2)) * 2;
        float w0 = W[o * CIN + c], w1 = W[o * CIN + c + 1];
        sW[o * LDK + c]     = (h16)w0;
        sW[o * LDK + c + 1] = (h16)w1;
    }
    // stage X with fused normalize + ReLU -> transposed f16 LDS
    for (int i2 = tid; i2 < CIN * (NTILE / 2); i2 += 128) {
        int c = i2 / (NTILE / 2);
        int j = (i2 - c * (NTILE / 2)) * 2;
        int p = pBase + j;
        float mean = inStats[256 + c], rstd = inStats[384 + c];
        float sc = rstd * inGamma[c], sh = inBeta[c] - mean * sc;
        float v0 = (float)actIn[c * P + p];
        float v1 = (float)actIn[c * P + p + 1];
        sXT[j * LDK + c]       = (h16)fmaxf(v0 * sc + sh, 0.0f);
        sXT[(j + 1) * LDK + c] = (h16)fmaxf(v1 * sc + sh, 0.0f);
    }
    __syncthreads();

    const int hi  = (lane >> 4) & 1;
    const int l15 = lane & 15;
    constexpr int MT = COUT / 16;
    constexpr int TILES = MT * (NTILE / 16);

#pragma unroll
    for (int t0 = 0; t0 < TILES; t0 += 4) {
        int t  = t0 + wv;
        int mt = t / (NTILE / 16);
        int nt = t - mt * (NTILE / 16);
        v8f acc = {0.f, 0.f, 0.f, 0.f, 0.f, 0.f, 0.f, 0.f};
#pragma unroll
        for (int kb = 0; kb < CIN; kb += 32) {
            // A 16x32 f16 layout: lane row = mt*16+l15; K runs {hi*8..hi*8+7}
            // and {16+hi*8..}; both 16B-aligned contiguous LDS runs.
            const h16* rowA = &sW[(mt * 16 + l15) * LDK + kb + hi * 8];
            union { v16h v; v8h h[2]; } au;
            au.h[0] = *(const v8h*)(rowA);
            au.h[1] = *(const v8h*)(rowA + 16);
            // B 32x16 f16 layout: lane col = nt*16+l15; K runs {hi*8..hi*8+7}
            // and {16+hi*8..}, matching A; two 16B runs in the transposed X tile.
            const h16* rowB = &sXT[(nt * 16 + l15) * LDK + kb + hi * 8];
            union { v16h v; v8h h[2]; } bu;
            bu.h[0] = *(const v8h*)(rowB);
            bu.h[1] = *(const v8h*)(rowB + 16);
            acc = __builtin_amdgcn_wmma_f32_16x16x32_f16(
                false, au.v, false, bu.v, (short)0, acc, false, false);
        }
        // C/D layout: VGPR i -> M = i + 8*hi ; N = lane&15
        int p = pBase + nt * 16 + l15;
        int base = (mt * 16 + hi * 8) * P + p;    // fits in int (<2^28)
#pragma unroll
        for (int i = 0; i < 8; ++i) {
            int ch = mt * 16 + hi * 8 + i;
            actOut[base + i * P] = (h16)(acc[i] + bias[ch]);
        }
    }
}

// ----------------------------------------------- maxpool + final norm ----
__global__ void maxpool_kernel(const h16* __restrict__ act, const float* __restrict__ stats,
                               const float* __restrict__ gamma, const float* __restrict__ beta,
                               float* __restrict__ out, int K, int C, int P, int chOff)
{
    int idx = blockIdx.x * blockDim.x + threadIdx.x;   // over C * (B*S), g fastest
    if (idx >= C * BATCH * SOUT) return;
    int c = idx / (BATCH * SOUT);
    int g = idx - c * (BATCH * SOUT);
    int b = g / SOUT, s = g - b * SOUT;
    float mean = stats[256 + c], rstd = stats[384 + c];
    float sc = rstd * gamma[c], sh = beta[c] - mean * sc;
    float m = 0.0f;                                    // relu output >= 0
    const v8h* a8 = (const v8h*)(act + c * P + g * K); // K multiple of 8, 16B aligned
    for (int k8 = 0; k8 < K / 8; ++k8) {
        v8h v = a8[k8];
#pragma unroll
        for (int e = 0; e < 8; ++e)
            m = fmaxf(m, fmaxf((float)v[e] * sc + sh, 0.0f));
    }
    out[b * 320 * SOUT + (chOff + c) * SOUT + s] = m;
}

// ------------------------------------------------------------- launch ----
extern "C" void kernel_launch(void* const* d_in, const int* in_sizes, int n_in,
                              void* d_out, int out_size, void* d_ws, size_t ws_size,
                              hipStream_t stream)
{
    (void)in_sizes; (void)n_in; (void)out_size; (void)ws_size;
    const float* xyz = (const float*)d_in[0];
    const float* pts = (const float*)d_in[1];
    // params pytree order per (branch,layer): W, b, beta, gamma (sorted dict keys)
    auto Wp  = [&](int br, int ly) { return (const float*)d_in[2 + (br * 3 + ly) * 4 + 0]; };
    auto Bp  = [&](int br, int ly) { return (const float*)d_in[2 + (br * 3 + ly) * 4 + 1]; };
    auto Bep = [&](int br, int ly) { return (const float*)d_in[2 + (br * 3 + ly) * 4 + 2]; };
    auto Gp  = [&](int br, int ly) { return (const float*)d_in[2 + (br * 3 + ly) * 4 + 3]; };

    char* ws = (char*)d_ws;
    size_t off = 0;
    int*   cent   = (int*)(ws + off);   off += (size_t)BATCH * SOUT * 4;
    float* newxyz = (float*)(ws + off); off += (size_t)BATCH * SOUT * 3 * 4;
    int*   g0     = (int*)(ws + off);   off += (size_t)BATCH * SOUT * 16 * 4;
    int*   g1     = (int*)(ws + off);   off += (size_t)BATCH * SOUT * 32 * 4;
    int*   g2     = (int*)(ws + off);   off += (size_t)BATCH * SOUT * 128 * 4;
    float* stats  = (float*)(ws + off); off += (size_t)9 * 512 * 4;
    off = (off + 255) & ~(size_t)255;
    const size_t ACTMAX = (size_t)BATCH * SOUT * 128 * 128;  // P_max * C_max elems
    h16* actA = (h16*)(ws + off); off += ACTMAX * sizeof(h16);
    h16* actB = (h16*)(ws + off); off += ACTMAX * sizeof(h16);

    float* out      = (float*)d_out;
    float* out_feat = out + (size_t)BATCH * 3 * SOUT;

    auto statsL = [&](int br, int ly) { return stats + (size_t)(br * 3 + ly) * 512; };
    auto run_stats = [&](const h16* act, float* st, int C, int P) {
        int bpc = (P + 32767) / 32768;
        stats_kernel<<<C * bpc, 256, 0, stream>>>(act, st, P, bpc);
        finalize_kernel<<<1, 128, 0, stream>>>(st, C, 1.0f / (float)P);
    };

    zero_stats_kernel<<<4, 256, 0, stream>>>(stats, 9 * 512);
    fps_kernel<<<BATCH, 256, 0, stream>>>(xyz, cent, newxyz, out);
    ballquery_kernel<<<(BATCH * SOUT) / 8, 256, 0, stream>>>(xyz, newxyz, g0, g1, g2);

    int* gidx[3] = {g0, g1, g2};
    int chOff = 0;

#define RUN_BRANCH(BR, KK, C0, C1, C2)                                                       \
    do {                                                                                     \
        const int P = BATCH * SOUT * (KK);                                                   \
        gather_l0_kernel<<<(P + 255) / 256, 256, 0, stream>>>(                               \
            xyz, pts, newxyz, gidx[BR], Wp(BR, 0), Bp(BR, 0), actA, (KK), (C0), P);          \
        run_stats(actA, statsL(BR, 0), (C0), P);                                             \
        mlp_wmma_kernel<C0, C1><<<P / NTILE, 128, 0, stream>>>(                              \
            actA, actB, Wp(BR, 1), Bp(BR, 1), statsL(BR, 0), Gp(BR, 0), Bep(BR, 0), P);      \
        run_stats(actB, statsL(BR, 1), (C1), P);                                             \
        mlp_wmma_kernel<C1, C2><<<P / NTILE, 128, 0, stream>>>(                              \
            actB, actA, Wp(BR, 2), Bp(BR, 2), statsL(BR, 1), Gp(BR, 1), Bep(BR, 1), P);      \
        run_stats(actA, statsL(BR, 2), (C2), P);                                             \
        maxpool_kernel<<<((C2) * BATCH * SOUT + 255) / 256, 256, 0, stream>>>(               \
            actA, statsL(BR, 2), Gp(BR, 2), Bep(BR, 2), out_feat, (KK), (C2), P, chOff);     \
        chOff += (C2);                                                                       \
    } while (0)

    RUN_BRANCH(0, 16, 32, 32, 64);
    RUN_BRANCH(1, 32, 64, 64, 128);
    RUN_BRANCH(2, 128, 64, 96, 128);
#undef RUN_BRANCH
}
